// myAttention_18013092839564
// MI455X (gfx1250) — matrix operand order repaired and checked by reading, not compiled, not run on hardware
//
#include <hip/hip_runtime.h>
#include <hip/hip_bf16.h>

// ---------------- problem constants ----------------
#define B_     64
#define N_     577
#define C_     768
#define H_     12
#define D_     64
#define KEEP_  289            // target_len = 577 - int(577*0.5)
#define MASKN_ 576
#define DROPK_ 288
#define KPAD_  320            // keys padded to 10 chunks of 32
#define MTOT_  (B_ * N_)      // 36928
#define QKVN_  (3 * C_)       // 2304

typedef __attribute__((ext_vector_type(16))) __bf16 v16bf;
typedef __attribute__((ext_vector_type(8)))  float  v8f;

__device__ __forceinline__ __bf16 f2bf(float f) {
    unsigned u = __builtin_bit_cast(unsigned, f);
    u += 0x7FFFu + ((u >> 16) & 1u);            // RNE
    unsigned short s = (unsigned short)(u >> 16);
    return __builtin_bit_cast(__bf16, s);
}
// A-matrix (16-bit, 16x32) per-lane K index for element e (contiguous: k = 16*half + e)
__device__ __forceinline__ int a_kidx(int e, int half) {
    return e + half * 16;
}

// ---------------- fp32 -> bf16 convert ----------------
__global__ __launch_bounds__(256)
void cvt_bf16_kernel(const float* __restrict__ src, __bf16* __restrict__ dst, int n) {
    int i = blockIdx.x * 256 + threadIdx.x;
    int stride = gridDim.x * 256;
    for (; i < n; i += stride) dst[i] = f2bf(src[i]);
}

// fp32 [K][Nn] -> bf16 transposed [Nn][K] (weights, done once per launch)
__global__ __launch_bounds__(256)
void cvt_transpose_kernel(const float* __restrict__ src, __bf16* __restrict__ dst,
                          int K, int Nn) {
    int idx = blockIdx.x * 256 + threadIdx.x;
    int total = K * Nn;
    int stride = gridDim.x * 256;
    for (; idx < total; idx += stride) {
        int k = idx / Nn, n = idx % Nn;
        dst[(size_t)n * K + k] = f2bf(src[idx]);
    }
}

// ---------------- token selection ----------------
// pos[bh*577 + n] = row in compacted K/V (0..288) or -1 if dropped.
__global__ __launch_bounds__(256)
void select_tokens_kernel(const float* __restrict__ mask, int* __restrict__ pos) {
    __shared__ float sv[MASKN_];
    __shared__ int   sk[MASKN_];
    const int bh  = blockIdx.x;
    const int tid = threadIdx.x;
    const float* mrow = mask + (size_t)bh * MASKN_;
    for (int i = tid; i < MASKN_; i += 256) sv[i] = mrow[i];
    __syncthreads();
    for (int i = tid; i < MASKN_; i += 256) {
        float v = sv[i];
        int rank = 0;
        for (int j = 0; j < MASKN_; ++j) {
            float w = sv[j];
            rank += (w > v) || (w == v && j < i);
        }
        sk[i] = (rank < DROPK_) ? 1 : 0;
    }
    __syncthreads();
    for (int i = tid; i < MASKN_; i += 256) {
        int p = 0;
        for (int j = 0; j < i; ++j) p += sk[j];
        pos[(size_t)bh * N_ + 1 + i] = sk[i] ? (1 + p) : -1;
    }
    if (tid == 0) pos[(size_t)bh * N_] = 0;
}

// ---------------- zero K/V padding rows ----------------
// kpb: [bh][KPAD_][D_] ; vpb: [bh][D_][KPAD_]
__global__ __launch_bounds__(256)
void zero_pad_kernel(__bf16* __restrict__ kpb, __bf16* __restrict__ vpb) {
    const int PADR = KPAD_ - KEEP_;
    const int total = B_ * H_ * PADR * D_;
    int idx = blockIdx.x * 256 + threadIdx.x;
    if (idx < total) {
        int d   = idx & (D_ - 1);
        int rr  = idx >> 6;
        int row = rr % PADR;
        int bh  = rr / PADR;
        __bf16 z = __builtin_bit_cast(__bf16, (unsigned short)0);
        kpb[((size_t)bh * KPAD_ + KEEP_ + row) * D_ + d] = z;
        vpb[((size_t)bh * D_ + d) * KPAD_ + KEEP_ + row] = z;
    }
}

// ---------------- GEMM1: qkv = x @ Wqkv + b, scatter q / gather k,v --------
// A [MTOT_][768] row-major, BwT [2304][768] (pre-transposed weight).
// block tile 128x128, 8 waves x (32x64), K-step 32.
__global__ __launch_bounds__(256)
void gemm_qkv_kernel(const __bf16* __restrict__ A,
                     const __bf16* __restrict__ BwT,   // [QKVN_][C_]
                     const float*  __restrict__ bias,
                     const int*    __restrict__ pos,
                     __bf16* __restrict__ qb,          // [bh][N_][D_]
                     __bf16* __restrict__ kpb,         // [bh][KPAD_][D_]
                     __bf16* __restrict__ vpb) {       // [bh][D_][KPAD_]
    __shared__ __bf16 As[128][32];
    __shared__ __bf16 Bt[128][32];                     // n-major!
    const int tid  = threadIdx.x;
    const int lane = tid & 31, wave = tid >> 5;
    const int half = lane >> 4, l16 = lane & 15;
    const int wr = (wave >> 1) * 32;
    const int wc = (wave & 1) * 64;
    const int bm = blockIdx.y * 128;
    const int bn = blockIdx.x * 128;

    v8f z = {};
    v8f acc[2][4];
#pragma unroll
    for (int i = 0; i < 2; ++i)
#pragma unroll
        for (int j = 0; j < 4; ++j) acc[i][j] = z;

    for (int k0 = 0; k0 < C_; k0 += 32) {
#pragma unroll
        for (int it = 0; it < 2; ++it) {               // A tile 128x32, 512 x uint4
            int g = it * 256 + tid;
            int r = g >> 2, c8 = (g & 3) * 8;
            int grow = bm + r; grow = grow < MTOT_ ? grow : MTOT_ - 1;  // clamp
            *(uint4*)(&As[r][c8]) =
                *(const uint4*)(A + (size_t)grow * C_ + k0 + c8);
        }
#pragma unroll
        for (int it = 0; it < 2; ++it) {               // B tile (n-major) 128x32
            int g = it * 256 + tid;
            int r = g >> 2, c8 = (g & 3) * 8;
            *(uint4*)(&Bt[r][c8]) =
                *(const uint4*)(BwT + (size_t)(bn + r) * C_ + k0 + c8);
        }
        if (k0 + 32 < C_) {
            __builtin_prefetch(A + (size_t)(bm + (tid & 127)) * C_ + k0 + 32, 0, 1);
            __builtin_prefetch(BwT + (size_t)(bn + (tid & 127)) * C_ + k0 + 32, 0, 1);
        }
        __syncthreads();

        v16bf af[2], bfm[4];
#pragma unroll
        for (int i = 0; i < 2; ++i)
#pragma unroll
            for (int e = 0; e < 16; ++e)
                af[i][e] = As[wr + i * 16 + l16][a_kidx(e, half)];
#pragma unroll
        for (int j = 0; j < 4; ++j)
#pragma unroll
            for (int e = 0; e < 16; ++e)                // contiguous per lane
                bfm[j][e] = Bt[wc + j * 16 + l16][e + 16 * half];
#pragma unroll
        for (int i = 0; i < 2; ++i)
#pragma unroll
            for (int j = 0; j < 4; ++j)
                acc[i][j] = __builtin_amdgcn_wmma_f32_16x16x32_bf16(
                    false, af[i], false, bfm[j], (short)0, acc[i][j], false, false);
        __syncthreads();
    }

#pragma unroll
    for (int i = 0; i < 2; ++i)
#pragma unroll
        for (int j = 0; j < 4; ++j)
#pragma unroll
            for (int r = 0; r < 8; ++r) {
                int m = bm + wr + i * 16 + r + half * 8;
                int n = bn + wc + j * 16 + l16;
                if (m < MTOT_) {
                    float v = acc[i][j][r] + bias[n];
                    int b   = m / N_, tok = m % N_;
                    int g   = n / C_, rem = n % C_;
                    int h   = rem >> 6, d = rem & 63;
                    int bh  = b * H_ + h;
                    __bf16 bv = f2bf(v);
                    if (g == 0) {
                        qb[((size_t)bh * N_ + tok) * D_ + d] = bv;
                    } else {
                        int p = pos[(size_t)bh * N_ + tok];
                        if (p >= 0) {
                            if (g == 1)
                                kpb[((size_t)bh * KPAD_ + p) * D_ + d] = bv;
                            else
                                vpb[((size_t)bh * D_ + d) * KPAD_ + p] = bv;
                        }
                    }
                }
            }
}

// ---------------- flash attention over compacted keys ----------------
// grid (B*H, 5): 256 thr = 8 waves, each wave owns 16 q rows (128 rows/block).
__global__ __launch_bounds__(256)
void attn_kernel(const __bf16* __restrict__ qb,        // [bh][N_][D_]
                 const __bf16* __restrict__ kpb,       // [bh][KPAD_][D_]
                 const __bf16* __restrict__ vpt,       // [bh][D_][KPAD_]
                 __bf16* __restrict__ ob) {            // [B_][N_][C_]
    __shared__ __bf16 kt[32][64];                      // [key][d]
    __shared__ __bf16 vt[64][32];                      // [d][key]  (transposed)
    __shared__ __bf16 pb[8][16][32];
    const int tid  = threadIdx.x;
    const int lane = tid & 31, wave = tid >> 5;
    const int half = lane >> 4, l16 = lane & 15;
    const int bh = blockIdx.x;
    const int m0 = blockIdx.y * 128 + wave * 16;
    const int b  = bh / H_, h = bh % H_;
    const float scale = 0.125f;

    v16bf qa[2];
#pragma unroll
    for (int s = 0; s < 2; ++s)
#pragma unroll
        for (int e = 0; e < 16; ++e) {
            int d   = a_kidx(e, half) + s * 32;
            int row = m0 + l16; row = row < N_ ? row : N_ - 1;   // clamp
            qa[s][e] = qb[((size_t)bh * N_ + row) * D_ + d];
        }

    v8f z = {};
    v8f oacc[4];
#pragma unroll
    for (int t = 0; t < 4; ++t) oacc[t] = z;
    float mrun[8], lrun[8];
#pragma unroll
    for (int r = 0; r < 8; ++r) { mrun[r] = -1e30f; lrun[r] = 0.f; }

    for (int ch = 0; ch < KPAD_ / 32; ++ch) {
        const int key0 = ch * 32;
        {   // stage K chunk [32][64]: 256 x uint4
            int r = tid >> 3, c8 = (tid & 7) * 8;
            *(uint4*)(&kt[r][c8]) =
                *(const uint4*)(kpb + ((size_t)bh * KPAD_ + key0 + r) * D_ + c8);
            // stage V chunk transposed [64][32]: 256 x uint4
            int rv = tid >> 2, cv8 = (tid & 3) * 8;
            *(uint4*)(&vt[rv][cv8]) =
                *(const uint4*)(vpt + ((size_t)bh * D_ + rv) * KPAD_ + key0 + cv8);
        }
        __syncthreads();

        v8f s0 = z, s1 = z;
#pragma unroll
        for (int st = 0; st < 2; ++st) {
            v16bf b0, b1;
#pragma unroll
            for (int e = 0; e < 16; ++e) {             // contiguous per lane
                int d = st * 32 + e + 16 * half;
                b0[e] = kt[l16][d];
                b1[e] = kt[16 + l16][d];
            }
            s0 = __builtin_amdgcn_wmma_f32_16x16x32_bf16(false, qa[st], false, b0,
                                                         (short)0, s0, false, false);
            s1 = __builtin_amdgcn_wmma_f32_16x16x32_bf16(false, qa[st], false, b1,
                                                         (short)0, s1, false, false);
        }

        const bool msk0 = (key0 + l16) >= KEEP_;
        const bool msk1 = (key0 + 16 + l16) >= KEEP_;
#pragma unroll
        for (int r = 0; r < 8; ++r) {
            float v0 = msk0 ? -1e30f : s0[r] * scale;
            float v1 = msk1 ? -1e30f : s1[r] * scale;
            float mx = fmaxf(v0, v1);
            mx = fmaxf(mx, __shfl_xor(mx, 1));
            mx = fmaxf(mx, __shfl_xor(mx, 2));
            mx = fmaxf(mx, __shfl_xor(mx, 4));
            mx = fmaxf(mx, __shfl_xor(mx, 8));
            float mnew = fmaxf(mrun[r], mx);
            float fac  = __expf(mrun[r] - mnew);
            float p0   = __expf(v0 - mnew);
            float p1   = __expf(v1 - mnew);
            float rs   = p0 + p1;
            rs += __shfl_xor(rs, 1);
            rs += __shfl_xor(rs, 2);
            rs += __shfl_xor(rs, 4);
            rs += __shfl_xor(rs, 8);
            lrun[r] = lrun[r] * fac + rs;
            mrun[r] = mnew;
#pragma unroll
            for (int t = 0; t < 4; ++t) oacc[t][r] *= fac;
            int mrow = r + half * 8;
            pb[wave][mrow][l16]      = f2bf(p0);
            pb[wave][mrow][16 + l16] = f2bf(p1);
        }

        v16bf pa;
#pragma unroll
        for (int e = 0; e < 16; ++e)                   // contiguous per lane
            pa[e] = pb[wave][l16][a_kidx(e, half)];
#pragma unroll
        for (int t = 0; t < 4; ++t) {
            v16bf bv;
#pragma unroll
            for (int e = 0; e < 16; ++e)               // contiguous per lane now
                bv[e] = vt[t * 16 + l16][e + 16 * half];
            oacc[t] = __builtin_amdgcn_wmma_f32_16x16x32_bf16(
                false, pa, false, bv, (short)0, oacc[t], false, false);
        }
        __syncthreads();
    }

#pragma unroll
    for (int r = 0; r < 8; ++r) {
        float inv = 1.f / lrun[r];
        int tok = m0 + r + half * 8;
        if (tok < N_) {
#pragma unroll
            for (int t = 0; t < 4; ++t)
                ob[((size_t)b * N_ + tok) * C_ + h * D_ + t * 16 + l16] =
                    f2bf(oacc[t][r] * inv);
        }
    }
}

// ---------------- GEMM2: out = attn @ Wproj + bproj (fp32 out) ------------
__global__ __launch_bounds__(256)
void gemm_proj_kernel(const __bf16* __restrict__ A,    // [MTOT_][C_]
                      const __bf16* __restrict__ BwT,  // [C_][C_] transposed
                      const float*  __restrict__ bias,
                      float* __restrict__ out) {
    __shared__ __bf16 As[128][32];
    __shared__ __bf16 Bt[128][32];
    const int tid  = threadIdx.x;
    const int lane = tid & 31, wave = tid >> 5;
    const int half = lane >> 4, l16 = lane & 15;
    const int wr = (wave >> 1) * 32;
    const int wc = (wave & 1) * 64;
    const int bm = blockIdx.y * 128;
    const int bn = blockIdx.x * 128;

    v8f z = {};
    v8f acc[2][4];
#pragma unroll
    for (int i = 0; i < 2; ++i)
#pragma unroll
        for (int j = 0; j < 4; ++j) acc[i][j] = z;

    for (int k0 = 0; k0 < C_; k0 += 32) {
#pragma unroll
        for (int it = 0; it < 2; ++it) {
            int g = it * 256 + tid;
            int r = g >> 2, c8 = (g & 3) * 8;
            int grow = bm + r; grow = grow < MTOT_ ? grow : MTOT_ - 1;
            *(uint4*)(&As[r][c8]) =
                *(const uint4*)(A + (size_t)grow * C_ + k0 + c8);
        }
#pragma unroll
        for (int it = 0; it < 2; ++it) {
            int g = it * 256 + tid;
            int r = g >> 2, c8 = (g & 3) * 8;
            *(uint4*)(&Bt[r][c8]) =
                *(const uint4*)(BwT + (size_t)(bn + r) * C_ + k0 + c8);
        }
        if (k0 + 32 < C_) {
            __builtin_prefetch(A + (size_t)(bm + (tid & 127)) * C_ + k0 + 32, 0, 1);
            __builtin_prefetch(BwT + (size_t)(bn + (tid & 127)) * C_ + k0 + 32, 0, 1);
        }
        __syncthreads();

        v16bf af[2], bfm[4];
#pragma unroll
        for (int i = 0; i < 2; ++i)
#pragma unroll
            for (int e = 0; e < 16; ++e)
                af[i][e] = As[wr + i * 16 + l16][a_kidx(e, half)];
#pragma unroll
        for (int j = 0; j < 4; ++j)
#pragma unroll
            for (int e = 0; e < 16; ++e)
                bfm[j][e] = Bt[wc + j * 16 + l16][e + 16 * half];
#pragma unroll
        for (int i = 0; i < 2; ++i)
#pragma unroll
            for (int j = 0; j < 4; ++j)
                acc[i][j] = __builtin_amdgcn_wmma_f32_16x16x32_bf16(
                    false, af[i], false, bfm[j], (short)0, acc[i][j], false, false);
        __syncthreads();
    }

#pragma unroll
    for (int i = 0; i < 2; ++i)
#pragma unroll
        for (int j = 0; j < 4; ++j)
#pragma unroll
            for (int r = 0; r < 8; ++r) {
                int m = bm + wr + i * 16 + r + half * 8;
                int n = bn + wc + j * 16 + l16;
                if (m < MTOT_)
                    out[(size_t)m * C_ + n] = acc[i][j][r] + bias[n];
            }
}

// ---------------- launcher ----------------
extern "C" void kernel_launch(void* const* d_in, const int* in_sizes, int n_in,
                              void* d_out, int out_size, void* d_ws, size_t ws_size,
                              hipStream_t stream) {
    (void)in_sizes; (void)n_in; (void)out_size; (void)ws_size;
    const float* x     = (const float*)d_in[0];
    const float* mask  = (const float*)d_in[1];
    const float* Wqkv  = (const float*)d_in[2];
    const float* bqkv  = (const float*)d_in[3];
    const float* Wproj = (const float*)d_in[4];
    const float* bproj = (const float*)d_in[5];
    float* out = (float*)d_out;

    auto alignup = [](size_t v) { return (v + 255) & ~(size_t)255; };
    char* w = (char*)d_ws;
    __bf16* x_bf  = (__bf16*)w; w += alignup(sizeof(__bf16) * (size_t)MTOT_ * C_);
    __bf16* wq_bf = (__bf16*)w; w += alignup(sizeof(__bf16) * (size_t)C_ * QKVN_);
    __bf16* wp_bf = (__bf16*)w; w += alignup(sizeof(__bf16) * (size_t)C_ * C_);
    __bf16* q_bf  = (__bf16*)w; w += alignup(sizeof(__bf16) * (size_t)B_ * H_ * N_ * D_);
    __bf16* kp_bf = (__bf16*)w; w += alignup(sizeof(__bf16) * (size_t)B_ * H_ * KPAD_ * D_);
    __bf16* vp_bf = (__bf16*)w; w += alignup(sizeof(__bf16) * (size_t)B_ * H_ * KPAD_ * D_);
    __bf16* o_bf  = (__bf16*)w; w += alignup(sizeof(__bf16) * (size_t)MTOT_ * C_);
    int*    pos   = (int*)w;    w += alignup(sizeof(int) * (size_t)B_ * H_ * N_);

    cvt_bf16_kernel<<<4096, 256, 0, stream>>>(x, x_bf, MTOT_ * C_);
    cvt_transpose_kernel<<<2048, 256, 0, stream>>>(Wqkv, wq_bf, C_, QKVN_);
    cvt_transpose_kernel<<<1024, 256, 0, stream>>>(Wproj, wp_bf, C_, C_);
    select_tokens_kernel<<<B_ * H_, 256, 0, stream>>>(mask, pos);
    {
        int total = B_ * H_ * (KPAD_ - KEEP_) * D_;
        zero_pad_kernel<<<(total + 255) / 256, 256, 0, stream>>>(kp_bf, vp_bf);
    }
    {
        dim3 grid(QKVN_ / 128, (MTOT_ + 127) / 128);
        gemm_qkv_kernel<<<grid, 256, 0, stream>>>(x_bf, wq_bf, bqkv, pos,
                                                  q_bf, kp_bf, vp_bf);
    }
    {
        dim3 grid(B_ * H_, (N_ + 127) / 128);
        attn_kernel<<<grid, 256, 0, stream>>>(q_bf, kp_bf, vp_bf, o_bf);
    }
    {
        dim3 grid(C_ / 128, (MTOT_ + 127) / 128);
        gemm_proj_kernel<<<grid, 256, 0, stream>>>(o_bf, wp_bf, bproj, out);
    }
}
